// DoubleALIFRNN_30571577213349
// MI455X (gfx1250) — matrix operand order repaired and checked by reading, compiled, not run on hardware
//
#include <hip/hip_runtime.h>
#include <hip/hip_bf16.h>

typedef __attribute__((ext_vector_type(16))) __bf16 v16bf;
typedef __attribute__((ext_vector_type(8)))  __bf16 v8bf;
typedef __attribute__((ext_vector_type(8)))  float  v8f;
typedef __attribute__((ext_vector_type(4)))  float  v4f;

#define T_STEPS 2048
#define BATCH   128
#define IN_F    256
#define H1N     512
#define H2N     512
#define ON      128
#define K1D     (IN_F + H1N)   /* 768  */
#define K2D     (H1N + H2N)    /* 1024 */
#define LDZ     520            /* padded LDS row stride (bf16 elems); 1040B % 16 == 0 */
#define B0C     0.01f
#define BETAC   1.8f

// ---------------------------------------------------------------------------
// WMMA helper: D = A(16x32 bf16) * B(32x16 bf16) + C(16x16 f32)
// ---------------------------------------------------------------------------
static __device__ __forceinline__ v8f wmma_bf16(v16bf a, v16bf b, v8f c) {
  return __builtin_amdgcn_wmma_f32_16x16x32_bf16(
      /*neg_a=*/false, a, /*neg_b=*/false, b,
      /*c_mod=*/(short)0, c, /*reuse_a=*/false, /*reuse_b=*/false);
}

static __device__ __forceinline__ void zero8(v8f& v) {
#pragma unroll
  for (int i = 0; i < 8; ++i) v[i] = 0.f;
}

// WGP-scope prefetch via inline asm (scope field 0 = WGP, which fills every cache
// level on a miss). __builtin_prefetch can only reach SE/SYS scope, which per
// the ISA prefetch table does NOT fill the WGP cache — pointless for our
// L2-resident weight stream. No memory clobber: prefetch has no arch effects.
static __device__ __forceinline__ void prefetch_wgp(const void* p) {
  asm volatile("global_prefetch_b8 %0, off" :: "v"(p));
}

// B fragment: lane holds W^T[n][kb .. kb+7] and W^T[n][kb+16 .. kb+23] (two b128 loads)
static __device__ __forceinline__ v16bf ld_bfrag(const __bf16* p) {
  v8bf lo = *(const v8bf*)(p);
  v8bf hi = *(const v8bf*)(p + 16);
  v16bf r;
#pragma unroll
  for (int i = 0; i < 8; ++i) { r[i] = lo[i]; r[i + 8] = hi[i]; }
  return r;
}

// A fragment from bf16 row-major LDS: elems 0..7 = K kb..kb+7, 8..15 = K kb+16..kb+23
static __device__ __forceinline__ v16bf ld_afrag_bf(const __bf16* p) {
  v8bf lo = *(const v8bf*)(p);
  v8bf hi = *(const v8bf*)(p + 16);
  v16bf r;
#pragma unroll
  for (int i = 0; i < 8; ++i) { r[i] = lo[i]; r[i + 8] = hi[i]; }
  return r;
}

// A fragment from f32 row-major global (x input), converted on the fly
static __device__ __forceinline__ v16bf ld_afrag_f32(const float* p) {
  v4f a0 = *(const v4f*)(p);
  v4f a1 = *(const v4f*)(p + 4);
  v4f b0 = *(const v4f*)(p + 16);
  v4f b1 = *(const v4f*)(p + 20);
  v16bf r;
#pragma unroll
  for (int i = 0; i < 4; ++i) {
    r[i]      = (__bf16)a0[i];
    r[i + 4]  = (__bf16)a1[i];
    r[i + 8]  = (__bf16)b0[i];
    r[i + 12] = (__bf16)b1[i];
  }
  return r;
}

// ---------------------------------------------------------------------------
// Prep kernel: convert weights f32 -> bf16, transposed to [N][K] so that
// WMMA B fragments are contiguous per lane.
// ---------------------------------------------------------------------------
__global__ void alif_wconvert(const float* __restrict__ W1,
                              const float* __restrict__ W2,
                              const float* __restrict__ Wo,
                              __bf16* __restrict__ W1t,
                              __bf16* __restrict__ W2t,
                              __bf16* __restrict__ Wot) {
  int idx = blockIdx.x * blockDim.x + threadIdx.x;
  const int N1 = K1D * H1N;   // 393216
  const int N2 = K2D * H2N;   // 524288
  const int N3 = H2N * ON;    // 65536
  if (idx < N1) {
    int n = idx % H1N, k = idx / H1N;
    W1t[(size_t)n * K1D + k] = (__bf16)W1[idx];
  } else if (idx < N1 + N2) {
    int j = idx - N1;
    int n = j % H2N, k = j / H2N;
    W2t[(size_t)n * K2D + k] = (__bf16)W2[j];
  } else if (idx < N1 + N2 + N3) {
    int j = idx - N1 - N2;
    int o = j % ON, k = j / ON;
    Wot[(size_t)o * H2N + k] = (__bf16)Wo[j];
  }
}

// ---------------------------------------------------------------------------
// Persistent recurrent kernel: 8 WGs x 16 batch rows, 8 waves x 64 hidden cols.
// The T=2048 recurrence runs entirely inside one launch; spikes are exchanged
// through LDS with 4 workgroup barriers per step.
// ---------------------------------------------------------------------------
__global__ __launch_bounds__(256)
void alif_persistent(const float* __restrict__ x,
                     const __bf16* __restrict__ W1t,
                     const __bf16* __restrict__ W2t,
                     const __bf16* __restrict__ Wot,
                     const float* __restrict__ tau_m1, const float* __restrict__ tau_a1,
                     const float* __restrict__ tau_m2, const float* __restrict__ tau_a2,
                     const float* __restrict__ tau_mo,
                     float* __restrict__ out) {
  __shared__ __align__(16) __bf16 z1s[16 * LDZ];
  __shared__ __align__(16) __bf16 z2s[16 * LDZ];

  const int tid  = threadIdx.x;
  const int wave = tid >> 5;
  const int lane = tid & 31;
  const int lm   = lane & 15;   // A row m / B,C column n (within tile)
  const int lhi  = lane >> 4;   // K-half / C row-half selector
  const int wg   = blockIdx.x;  // batch rows [wg*16, wg*16+16)

  // zero spike buffers (initial state z = 0)
  for (int i = tid; i < 16 * LDZ; i += 256) {
    z1s[i] = (__bf16)0.f;
    z2s[i] = (__bf16)0.f;
  }

  // per-column decay constants (column n owned by this lane per tile)
  float al1[4], rh1[4], al2[4], rh2[4];
#pragma unroll
  for (int tt = 0; tt < 4; ++tt) {
    int n = wave * 64 + tt * 16 + lm;
    al1[tt] = __expf(-1.f / tau_m1[n]);
    rh1[tt] = __expf(-1.f / tau_a1[n]);
    al2[tt] = __expf(-1.f / tau_m2[n]);
    rh2[tt] = __expf(-1.f / tau_a2[n]);
  }
  const int   no = wave * 16 + lm;
  const float ao = __expf(-1.f / tau_mo[no]);

  // persistent neuron state in C-matrix layout (lane: 8 rows x 1 col per tile)
  v8f u1[4], a1[4], u2[4], a2[4], uo;
#pragma unroll
  for (int tt = 0; tt < 4; ++tt) { zero8(u1[tt]); zero8(a1[tt]); zero8(u2[tt]); zero8(a2[tt]); }
  zero8(uo);

  // hoisted weight row bases for this wave/lane
  const __bf16* w1row[4];
  const __bf16* w2row[4];
#pragma unroll
  for (int tt = 0; tt < 4; ++tt) {
    int n = wave * 64 + tt * 16 + lm;
    w1row[tt] = W1t + (size_t)n * K1D + lhi * 8;
    w2row[tt] = W2t + (size_t)n * K2D + lhi * 8;
  }
  const __bf16* worow = Wot + (size_t)no * H2N + lhi * 8;

  __syncthreads();

  for (int t = 0; t < T_STEPS; ++t) {
    // ============================ layer 1 GEMM ============================
    v8f acc[4];
#pragma unroll
    for (int tt = 0; tt < 4; ++tt) zero8(acc[tt]);

    // input part: x_t[16,256] (f32 -> bf16 on the fly)
    const float* xrow = x + ((size_t)t * BATCH + wg * 16 + lm) * IN_F + lhi * 8;
#pragma unroll 2
    for (int k0 = 0; k0 < IN_F; k0 += 32) {
      v16bf af = ld_afrag_f32(xrow + k0);
#pragma unroll
      for (int tt = 0; tt < 4; ++tt) {
        prefetch_wgp(w1row[tt] + k0 + 32);
        acc[tt] = wmma_bf16(af, ld_bfrag(w1row[tt] + k0), acc[tt]);
      }
    }
    // recurrent part: z1_old[16,512] from LDS
#pragma unroll 2
    for (int k0 = 0; k0 < H1N; k0 += 32) {
      v16bf af = ld_afrag_bf(z1s + lm * LDZ + k0 + lhi * 8);
#pragma unroll
      for (int tt = 0; tt < 4; ++tt) {
        prefetch_wgp(w1row[tt] + IN_F + k0 + 32);
        acc[tt] = wmma_bf16(af, ld_bfrag(w1row[tt] + IN_F + k0), acc[tt]);
      }
    }

    // ============================ ALIF layer 1 ============================
    float zn[4][8];
#pragma unroll
    for (int tt = 0; tt < 4; ++tt) {
      int col = wave * 64 + tt * 16 + lm;
#pragma unroll
      for (int r = 0; r < 8; ++r) {
        int   row = r + 8 * lhi;
        float zold = (float)z1s[row * LDZ + col];
        float an   = rh1[tt] * a1[tt][r] + (1.f - rh1[tt]) * zold;
        float th   = B0C + BETAC * an;
        float un   = al1[tt] * u1[tt][r] + (1.f - al1[tt]) * acc[tt][r] - zold * th;
        a1[tt][r] = an;
        u1[tt][r] = un;
        zn[tt][r] = (un - th) > 0.f ? 1.f : 0.f;
      }
    }
    __syncthreads();   // all reads of z1_old done
#pragma unroll
    for (int tt = 0; tt < 4; ++tt) {
      int col = wave * 64 + tt * 16 + lm;
#pragma unroll
      for (int r = 0; r < 8; ++r)
        z1s[(r + 8 * lhi) * LDZ + col] = (__bf16)zn[tt][r];
    }
    __syncthreads();   // z1_new visible

    // ============================ layer 2 GEMM ============================
#pragma unroll
    for (int tt = 0; tt < 4; ++tt) zero8(acc[tt]);
#pragma unroll 2
    for (int k0 = 0; k0 < H1N; k0 += 32) {
      v16bf af = ld_afrag_bf(z1s + lm * LDZ + k0 + lhi * 8);   // z1_new
#pragma unroll
      for (int tt = 0; tt < 4; ++tt) {
        prefetch_wgp(w2row[tt] + k0 + 32);
        acc[tt] = wmma_bf16(af, ld_bfrag(w2row[tt] + k0), acc[tt]);
      }
    }
#pragma unroll 2
    for (int k0 = 0; k0 < H2N; k0 += 32) {
      v16bf af = ld_afrag_bf(z2s + lm * LDZ + k0 + lhi * 8);   // z2_old
#pragma unroll
      for (int tt = 0; tt < 4; ++tt) {
        prefetch_wgp(w2row[tt] + H1N + k0 + 32);
        acc[tt] = wmma_bf16(af, ld_bfrag(w2row[tt] + H1N + k0), acc[tt]);
      }
    }

    // ============================ ALIF layer 2 ============================
#pragma unroll
    for (int tt = 0; tt < 4; ++tt) {
      int col = wave * 64 + tt * 16 + lm;
#pragma unroll
      for (int r = 0; r < 8; ++r) {
        int   row = r + 8 * lhi;
        float zold = (float)z2s[row * LDZ + col];
        float an   = rh2[tt] * a2[tt][r] + (1.f - rh2[tt]) * zold;
        float th   = B0C + BETAC * an;
        float un   = al2[tt] * u2[tt][r] + (1.f - al2[tt]) * acc[tt][r] - zold * th;
        a2[tt][r] = an;
        u2[tt][r] = un;
        zn[tt][r] = (un - th) > 0.f ? 1.f : 0.f;
      }
    }
    __syncthreads();   // all reads of z2_old done
#pragma unroll
    for (int tt = 0; tt < 4; ++tt) {
      int col = wave * 64 + tt * 16 + lm;
#pragma unroll
      for (int r = 0; r < 8; ++r)
        z2s[(r + 8 * lhi) * LDZ + col] = (__bf16)zn[tt][r];
    }
    __syncthreads();   // z2_new visible

    // ======================= output layer (per-wave tile) =================
    v8f acc3;
    zero8(acc3);
#pragma unroll 2
    for (int k0 = 0; k0 < H2N; k0 += 32) {
      v16bf af = ld_afrag_bf(z2s + lm * LDZ + k0 + lhi * 8);   // z2_new
      prefetch_wgp(worow + k0 + 32);
      acc3 = wmma_bf16(af, ld_bfrag(worow + k0), acc3);
    }
    float* orow = out + (size_t)t * (BATCH * ON) + (size_t)(wg * 16 + 8 * lhi) * ON + no;
#pragma unroll
    for (int r = 0; r < 8; ++r) {
      float v = ao * uo[r] + (1.f - ao) * acc3[r];
      uo[r]   = v;
      orow[(size_t)r * ON] = v;
    }
  }

  // ============================ final states ============================
  const size_t F  = (size_t)T_STEPS * BATCH * ON;
  const size_t SH = (size_t)BATCH * H1N;   // 65536 (H1N == H2N)

  // z1f, z2f from LDS (cooperative copy, bf16 spikes are exact 0/1)
  for (int i = tid; i < 16 * H1N; i += 256) {
    int row = i / H1N, col = i % H1N;
    out[F + 0 * SH + (size_t)(wg * 16 + row) * H1N + col] = (float)z1s[row * LDZ + col];
    out[F + 3 * SH + (size_t)(wg * 16 + row) * H2N + col] = (float)z2s[row * LDZ + col];
  }
  // u/a states from registers
#pragma unroll
  for (int tt = 0; tt < 4; ++tt) {
    int col = wave * 64 + tt * 16 + lm;
#pragma unroll
    for (int r = 0; r < 8; ++r) {
      size_t row = (size_t)(wg * 16 + r + 8 * lhi);
      out[F + 1 * SH + row * H1N + col] = u1[tt][r];
      out[F + 2 * SH + row * H1N + col] = a1[tt][r];
      out[F + 4 * SH + row * H2N + col] = u2[tt][r];
      out[F + 5 * SH + row * H2N + col] = a2[tt][r];
    }
  }
#pragma unroll
  for (int r = 0; r < 8; ++r)
    out[F + 6 * SH + (size_t)(wg * 16 + r + 8 * lhi) * ON + no] = uo[r];
}

// ---------------------------------------------------------------------------
// Launch
// ---------------------------------------------------------------------------
extern "C" void kernel_launch(void* const* d_in, const int* in_sizes, int n_in,
                              void* d_out, int out_size, void* d_ws, size_t ws_size,
                              hipStream_t stream) {
  const float* x   = (const float*)d_in[0];
  const float* W1  = (const float*)d_in[1];
  const float* W2  = (const float*)d_in[2];
  const float* Wo  = (const float*)d_in[3];
  const float* tm1 = (const float*)d_in[4];
  const float* ta1 = (const float*)d_in[5];
  const float* tm2 = (const float*)d_in[6];
  const float* ta2 = (const float*)d_in[7];
  const float* tmo = (const float*)d_in[8];
  float* out = (float*)d_out;

  // workspace layout: bf16 transposed weights (~1.97 MB total)
  __bf16* W1t = (__bf16*)d_ws;                       // 768*512
  __bf16* W2t = W1t + (size_t)K1D * H1N;             // 1024*512
  __bf16* Wot = W2t + (size_t)K2D * H2N;             // 512*128

  const int total = K1D * H1N + K2D * H2N + H2N * ON;  // 983040
  alif_wconvert<<<dim3((total + 255) / 256), dim3(256), 0, stream>>>(
      W1, W2, Wo, W1t, W2t, Wot);

  alif_persistent<<<dim3(BATCH / 16), dim3(256), 0, stream>>>(
      x, W1t, W2t, Wot, tm1, ta1, tm2, ta2, tmo, out);
}
